// BoundaryDoULoss_21105469293161
// MI455X (gfx1250) — compile-verified, matrix operand order checked by reading, not run on hardware
//
#include <hip/hip_runtime.h>
#include <hip/hip_bf16.h>

// ---- problem constants (fixed by setup_inputs) ----
#define BDIM 8
#define CDIM 2
#define HDIM 1024
#define WDIM 1024
#define ROWS 8                      // rows per block
#define TILES_PER_SLICE (HDIM / ROWS)   // 128
#define NBLOCKS (BDIM * CDIM * TILES_PER_SLICE) // 2048
#define SMOOTHF 1e-5f

typedef __attribute__((ext_vector_type(4)))  float    fvec4;
typedef __attribute__((ext_vector_type(4)))  int      ivec4;
typedef __attribute__((ext_vector_type(16))) _Float16 v16h;
typedef __attribute__((ext_vector_type(8)))  float    v8f;

// ws layout: [class*4 + {0:S, 1:C, 2:I, 3:Z}]  (8 floats)
__global__ void bdou_zero_ws(float* ws) {
    if (threadIdx.x < 8) ws[threadIdx.x] = 0.0f;
}

__global__ __launch_bounds__(256)
void bdou_reduce(const float* __restrict__ pred,
                 const int*   __restrict__ target,
                 float* __restrict__ ws) {
    __shared__ int   s_t[(ROWS + 2) * WDIM];   // 40 KB target tile incl. halo rows
    __shared__ float s_red[32];                // 8 waves x 4 accumulators

    const int tid   = threadIdx.x;
    const int slice = blockIdx.x >> 7;          // b*CDIM + c
    const int tile  = blockIdx.x & (TILES_PER_SLICE - 1);
    const int cls   = slice & (CDIM - 1);
    const int r0    = tile * ROWS;
    const size_t sliceOff = (size_t)slice * HDIM * WDIM;
    const int x0 = tid * 4;

    // ---- async-fill target rows [r0-1, r0+ROWS] into LDS (CDNA5 ASYNC path) --
    // Each iteration: 256 lanes x b128 = exactly one 1024-wide row, moved
    // global->LDS directly (no VGPR round trip), tracked by ASYNCcnt.
#pragma unroll
    for (int i = 0; i < ROWS + 2; ++i) {
        const int gr = r0 - 1 + i;
        if (gr >= 0 && gr < HDIM) {
            const unsigned lds_off =
                (unsigned)(unsigned long long)(const void*)&s_t[i * WDIM + x0];
            const unsigned long long gaddr =
                (unsigned long long)&target[sliceOff + (size_t)gr * WDIM + x0];
            asm volatile("global_load_async_to_lds_b128 %0, %1, off"
                         :: "v"(lds_off), "v"(gaddr)
                         : "memory");
        } else {
            // y-halo outside the image: zero fill (only tiles 0 and 127)
            ivec4 z = {0, 0, 0, 0};
            *(ivec4*)&s_t[i * WDIM + x0] = z;
        }
    }

    // ---- overlap: prefetch all pred rows (single-use -> non-temporal) while
    //      the async LDS fill is in flight ----
    fvec4 pv[ROWS];
#pragma unroll
    for (int r = 0; r < ROWS; ++r) {
        pv[r] = __builtin_nontemporal_load(
            (const fvec4*)&pred[sliceOff + (size_t)(r0 + r) * WDIM + x0]);
    }

    asm volatile("s_wait_asynccnt 0x0" ::: "memory");
    __syncthreads();

    // ---- streaming stencil + partial sums (32 pixels / thread) ----
    float Sa = 0.f, Ca = 0.f, Ia = 0.f, Za = 0.f;
#pragma unroll
    for (int r = 0; r < ROWS; ++r) {
        const fvec4 p = pv[r];
        const int base = (r + 1) * WDIM + x0;
        const ivec4 tc = *(const ivec4*)&s_t[base];
        const ivec4 tu = *(const ivec4*)&s_t[base - WDIM];
        const ivec4 td = *(const ivec4*)&s_t[base + WDIM];
        const int tl = (x0 > 0)        ? s_t[base - 1] : 0;
        const int tr = (x0 < WDIM - 4) ? s_t[base + 4] : 0;

        // z_sum is unmasked: sum of pred^2 over all pixels
        Za += p[0]*p[0] + p[1]*p[1] + p[2]*p[2] + p[3]*p[3];

        const int lefts[4]  = { tl,    tc[0], tc[1], tc[2] };
        const int rights[4] = { tc[1], tc[2], tc[3], tr    };
#pragma unroll
        for (int q = 0; q < 4; ++q) {
            if (tc[q]) {                          // t==1 (values are 0/1)
                Sa += 1.0f;
                Ia += p[q];
                const int interior = tu[q] & td[q] & lefts[q] & rights[q];
                if (!interior) Ca += 1.0f;        // boundary pixel (Y!=5 path)
            }
        }
    }

    // ---- wave32 butterfly reduction ----
#pragma unroll
    for (int off = 16; off > 0; off >>= 1) {
        Sa += __shfl_xor(Sa, off);
        Ca += __shfl_xor(Ca, off);
        Ia += __shfl_xor(Ia, off);
        Za += __shfl_xor(Za, off);
    }
    if ((tid & 31) == 0) {
        const int w = tid >> 5;                  // 8 waves
        s_red[w * 4 + 0] = Sa;
        s_red[w * 4 + 1] = Ca;
        s_red[w * 4 + 2] = Ia;
        s_red[w * 4 + 3] = Za;
    }
    __syncthreads();

    // ---- cross-wave reduction with a single WMMA (B = ones => row sums) ----
    // A(16x32 f16) row m<4 holds the 8 wave-partials of accumulator m in K=0..7
    // (A layout: lanes 0-15 carry M=lane, halves 0..7 = K0..7). D = A*ones + 0
    // => lane 0, D-vgpr j == total sum of accumulator j.
    if (tid < 32) {
        const int l = tid;
        const int m = l & 15;
        v16h a;
        v16h b;
#pragma unroll
        for (int e = 0; e < 16; ++e) { a[e] = (_Float16)0.0f; b[e] = (_Float16)1.0f; }
        if (l < 16 && m < 4) {
#pragma unroll
            for (int w = 0; w < 8; ++w) a[w] = (_Float16)s_red[w * 4 + m];
        }
        v8f c = {0.f, 0.f, 0.f, 0.f, 0.f, 0.f, 0.f, 0.f};
        c = __builtin_amdgcn_wmma_f32_16x16x32_f16(
                /*neg_a=*/false, a, /*neg_b=*/false, b,
                /*c_mod=*/(short)0, c, /*reuse_a=*/false, /*reuse_b=*/false);
        if (l == 0) {
            atomicAdd(&ws[cls * 4 + 0], c[0]);   // S
            atomicAdd(&ws[cls * 4 + 1], c[1]);   // C (boundary count)
            atomicAdd(&ws[cls * 4 + 2], c[2]);   // intersect
            atomicAdd(&ws[cls * 4 + 3], c[3]);   // z_sum
        }
    }
}

__global__ void bdou_finalize(const float* __restrict__ ws,
                              float* __restrict__ out) {
    if (threadIdx.x != 0 || blockIdx.x != 0) return;
    float total = 0.0f;
#pragma unroll
    for (int c = 0; c < CDIM; ++c) {
        const float S  = ws[c * 4 + 0];
        const float Cb = ws[c * 4 + 1];
        const float I  = ws[c * 4 + 2];
        const float Z  = ws[c * 4 + 3];
        float alpha = 1.0f - (Cb + SMOOTHF) / (S + SMOOTHF);
        alpha = 2.0f * alpha - 1.0f;
        alpha = fminf(alpha, 0.8f);
        const float y_sum = S;  // t*t == t for 0/1 targets
        const float loss = (Z + y_sum - 2.0f * I + SMOOTHF) /
                           (Z + y_sum - (1.0f + alpha) * I + SMOOTHF);
        total += loss;          // LOSS_WEIGHT == 1.0
    }
    out[0] = total / (float)CDIM;
}

extern "C" void kernel_launch(void* const* d_in, const int* in_sizes, int n_in,
                              void* d_out, int out_size, void* d_ws, size_t ws_size,
                              hipStream_t stream) {
    const float* pred   = (const float*)d_in[0];
    const int*   target = (const int*)d_in[1];
    float* out = (float*)d_out;
    float* ws  = (float*)d_ws;

    bdou_zero_ws<<<1, 32, 0, stream>>>(ws);
    bdou_reduce<<<NBLOCKS, 256, 0, stream>>>(pred, target, ws);
    bdou_finalize<<<1, 1, 0, stream>>>(ws, out);
}
